// Seq2SeqLSTM_29738353557713
// MI455X (gfx1250) — matrix operand order repaired and checked by reading, not compiled, not run on hardware
//
#include <hip/hip_runtime.h>

namespace {
constexpr int H   = 64;
constexpr int F   = 8;
constexpr int P   = 64;
constexpr int Bsz = 1024;
constexpr int T   = 512;
constexpr int BT  = 16;             // batch rows per workgroup (one WMMA M-tile)
constexpr int NW  = 4;              // waves per workgroup (16 hidden columns each)
constexpr int NT  = NW * 32;        // 128 threads

typedef __attribute__((ext_vector_type(16))) __bf16 v16bf;
typedef __attribute__((ext_vector_type(8)))  float  v8f;

struct SmemT {
  __bf16 hb[2][BT][H];    // ping-pong bf16 hidden state (WMMA A source)
  __bf16 xb[2][BT][32];   // ping-pong bf16 input tile, K padded 8 -> 32 with zeros
  float  h32[BT][H];      // f32 hidden state (decoder FC head only)
  float  fcw[F][H];
  float  fcb[F];
};

// Branch-free transcendentals on the recurrence critical path.
#if __has_builtin(__builtin_amdgcn_tanhf)
__device__ __forceinline__ float ftanh(float x) { return __builtin_amdgcn_tanhf(x); }
#else
__device__ __forceinline__ float ftanh(float x) {
  // tanh(x) = 2/(1+exp2(-2*log2e*x)) - 1 ; saturates correctly at +-1
  return 2.0f * __builtin_amdgcn_rcpf(1.0f + __builtin_amdgcn_exp2f(x * -2.8853900817779268f)) - 1.0f;
}
#endif
__device__ __forceinline__ float fsigm(float x) {
  return __builtin_amdgcn_rcpf(1.0f + __builtin_amdgcn_exp2f(x * -1.4426950408889634f));
}

// B-matrix (K x 16) lane fragment from row-major W[n][k]; w_row = &W[n][0].
// Lanes 0-15: elems 0-7 -> K=kb+0..7, elems 8-15 -> K=kb+16..23; lanes 16-31: K=kb+8..15, kb+24..31.
__device__ __forceinline__ v16bf b_frag_f32(const float* w_row, int kb, int kmax, int lane) {
  const int base = kb + ((lane & 16) ? 8 : 0);
  v16bf r;
#pragma unroll
  for (int i = 0; i < 16; ++i) {
    const int k = base + i + ((i < 8) ? 0 : 8);
    r[i] = (__bf16)((k < kmax) ? w_row[k] : 0.0f);
  }
  return r;
}

// A-matrix (16 x 32) lane fragment from row-major bf16 LDS tile (leading dim ld).
// Lane m (0-15): halves 0-7 -> K=kb+0..7, halves 8-15 -> K=kb+16..23;
// lane m+16:     halves 0-7 -> K=kb+8..15, halves 8-15 -> K=kb+24..31.
__device__ __forceinline__ v16bf a_frag_lds(const __bf16* src, int ld, int kb, int lane) {
  const int m   = lane & 15;
  const int off = (lane & 16) ? 8 : 0;
  v16bf r;
#pragma unroll
  for (int i = 0; i < 16; ++i) {
    const int k = kb + i + ((i < 8) ? 0 : 8) + off;
    r[i] = src[m * ld + k];
  }
  return r;
}
} // namespace

__global__ __launch_bounds__(NT) void seq2seq_lstm_kernel(
    const float* __restrict__ x_in,
    const float* __restrict__ eWih, const float* __restrict__ eWhh,
    const float* __restrict__ eBih, const float* __restrict__ eBhh,
    const float* __restrict__ dWih, const float* __restrict__ dWhh,
    const float* __restrict__ dBih, const float* __restrict__ dBhh,
    const float* __restrict__ fcW,  const float* __restrict__ fcB,
    float* __restrict__ out) {
  __shared__ SmemT s;
  const int tid  = threadIdx.x;
  const int wave = tid >> 5;
  const int lane = tid & 31;
  const int m16  = lane & 15;
  const int rb   = (lane & 16) ? 8 : 0;   // C/D row base for hi half-wave
  const int row0 = blockIdx.x * BT;
  const int col  = wave * 16 + m16;       // hidden column this lane owns (all 4 gates)

  // ---- one-time init -------------------------------------------------------
  for (int i = tid; i < 2 * BT * H; i += NT)  (&s.hb[0][0][0])[i] = (__bf16)0.0f;
  for (int i = tid; i < 2 * BT * 32; i += NT) (&s.xb[0][0][0])[i] = (__bf16)0.0f;
  for (int i = tid; i < F * H; i += NT)       (&s.fcw[0][0])[i] = fcW[i];
  if (tid < F) s.fcb[tid] = fcB[tid];

  // cell state: lane owns (rows rb..rb+7, column `col`) -- same mapping both phases
  float c_reg[8];
#pragma unroll
  for (int r = 0; r < 8; ++r) c_reg[r] = 0.0f;

  // Weight fragments + per-lane gate biases pinned in VGPRs for a whole phase.
  // Gate g of this lane's column lives at output column n = g*64 + col.
  v16bf bh0[4], bh1[4], bxw[4];
  float bias[4];

  // ---- encoder weights -----------------------------------------------------
#pragma unroll
  for (int g = 0; g < 4; ++g) {
    const int n = g * 64 + col;
    bh0[g] = b_frag_f32(eWhh + n * H, 0,  H, lane);
    bh1[g] = b_frag_f32(eWhh + n * H, 32, H, lane);
    bxw[g] = b_frag_f32(eWih + n * F, 0,  F, lane);
    bias[g] = eBih[n] + eBhh[n];
  }

  // x staging mapping + software-pipelined global load
  const int    xrow  = tid >> 3;
  const int    xf    = tid & 7;
  const size_t xbase = (size_t)(row0 + xrow) * (T * F) + xf;
  float xval = x_in[xbase];   // t = 0

  // ---- encoder: 1 barrier per step ----------------------------------------
  for (int t = 0; t < T; ++t) {
    s.xb[t & 1][xrow][xf] = (__bf16)xval;
    if (t + 1 < T) xval = x_in[xbase + (size_t)(t + 1) * F];   // prefetch next step
    __syncthreads();   // xb[t&1] + hb[t&1] (prev step's writes) visible

    const v16bf a0 = a_frag_lds(&s.hb[t & 1][0][0], H, 0,  lane);
    const v16bf a1 = a_frag_lds(&s.hb[t & 1][0][0], H, 32, lane);
    const v16bf ax = a_frag_lds(&s.xb[t & 1][0][0], 32, 0, lane);

    v8f acc[4];
#pragma unroll
    for (int g = 0; g < 4; ++g) {
      v8f a;
#pragma unroll
      for (int r = 0; r < 8; ++r) a[r] = bias[g];   // bias folded into C
      a = __builtin_amdgcn_wmma_f32_16x16x32_bf16(false, a0, false, bh0[g], (short)0, a, false, false);
      a = __builtin_amdgcn_wmma_f32_16x16x32_bf16(false, a1, false, bh1[g], (short)0, a, false, false);
      a = __builtin_amdgcn_wmma_f32_16x16x32_bf16(false, ax, false, bxw[g], (short)0, a, false, false);
      acc[g] = a;
    }

    // LSTM update entirely in registers: acc[0]=i, acc[1]=f, acc[2]=g, acc[3]=o
#pragma unroll
    for (int r = 0; r < 8; ++r) {
      const float cn = fsigm(acc[1][r]) * c_reg[r] + fsigm(acc[0][r]) * ftanh(acc[2][r]);
      c_reg[r] = cn;
      const float hn = fsigm(acc[3][r]) * ftanh(cn);
      s.hb[(t + 1) & 1][rb + r][col] = (__bf16)hn;
    }
  }
  // After encoder: hidden state in hb[0] (T even), x_last staged in xb[1].

  // ---- decoder weights -----------------------------------------------------
#pragma unroll
  for (int g = 0; g < 4; ++g) {
    const int n = g * 64 + col;
    bh0[g] = b_frag_f32(dWhh + n * H, 0,  H, lane);
    bh1[g] = b_frag_f32(dWhh + n * H, 32, H, lane);
    bxw[g] = b_frag_f32(dWih + n * F, 0,  F, lane);
    bias[g] = dBih[n] + dBhh[n];
  }

  // ---- decoder: 2 barriers per step ---------------------------------------
  for (int p = 0; p < P; ++p) {
    __syncthreads();   // hb[p&1] + xb[(p+1)&1] writes visible

    const v16bf a0 = a_frag_lds(&s.hb[p & 1][0][0], H, 0,  lane);
    const v16bf a1 = a_frag_lds(&s.hb[p & 1][0][0], H, 32, lane);
    const v16bf ax = a_frag_lds(&s.xb[(p + 1) & 1][0][0], 32, 0, lane);

    v8f acc[4];
#pragma unroll
    for (int g = 0; g < 4; ++g) {
      v8f a;
#pragma unroll
      for (int r = 0; r < 8; ++r) a[r] = bias[g];
      a = __builtin_amdgcn_wmma_f32_16x16x32_bf16(false, a0, false, bh0[g], (short)0, a, false, false);
      a = __builtin_amdgcn_wmma_f32_16x16x32_bf16(false, a1, false, bh1[g], (short)0, a, false, false);
      a = __builtin_amdgcn_wmma_f32_16x16x32_bf16(false, ax, false, bxw[g], (short)0, a, false, false);
      acc[g] = a;
    }

#pragma unroll
    for (int r = 0; r < 8; ++r) {
      const float cn = fsigm(acc[1][r]) * c_reg[r] + fsigm(acc[0][r]) * ftanh(acc[2][r]);
      c_reg[r] = cn;
      const float hn = fsigm(acc[3][r]) * ftanh(cn);
      s.hb[(p + 1) & 1][rb + r][col] = (__bf16)hn;
      s.h32[rb + r][col] = hn;
    }
    __syncthreads();   // h32 visible for FC head

    // FC head: pred = h @ fcW^T + fcb ; one output element per thread
    float pacc = s.fcb[xf];
#pragma unroll
    for (int k = 0; k < H; ++k) pacc += s.h32[xrow][k] * s.fcw[xf][k];
    out[(size_t)(row0 + xrow) * (P * F) + (size_t)p * F + xf] = pacc;
    s.xb[p & 1][xrow][xf] = (__bf16)pacc;   // autoregressive feedback (next step's input)
  }
}

extern "C" void kernel_launch(void* const* d_in, const int* in_sizes, int n_in,
                              void* d_out, int out_size, void* d_ws, size_t ws_size,
                              hipStream_t stream) {
  (void)in_sizes; (void)n_in; (void)out_size; (void)d_ws; (void)ws_size;
  const float* x_in = (const float*)d_in[0];
  const float* eWih = (const float*)d_in[1];
  const float* eWhh = (const float*)d_in[2];
  const float* eBih = (const float*)d_in[3];
  const float* eBhh = (const float*)d_in[4];
  const float* dWih = (const float*)d_in[5];
  const float* dWhh = (const float*)d_in[6];
  const float* dBih = (const float*)d_in[7];
  const float* dBhh = (const float*)d_in[8];
  const float* fcW  = (const float*)d_in[9];
  const float* fcB  = (const float*)d_in[10];
  float* out = (float*)d_out;

  seq2seq_lstm_kernel<<<Bsz / BT, NT, 0, stream>>>(
      x_in, eWih, eWhh, eBih, eBhh, dWih, dWhh, dBih, dBhh, fcW, fcB, out);
}
